// RelPosSelfAttention_22840636080185
// MI455X (gfx1250) — matrix operand order repaired and checked by reading, not compiled, not run on hardware
//
#include <hip/hip_runtime.h>
#include <hip/hip_bf16.h>
#include <math.h>

// ---------------------------------------------------------------------------
// RelPos self-attention for MI455X (gfx1250, wave32, WMMA bf16 16x16x32).
//
// B=4, N=1024, D=512, H=8, dh=64, HID=32.
// Pipeline: cast/transpose -> QKV GEMM (WMMA) -> fused bias-MLP + flash
// attention (WMMA + double-buffered GLOBAL_LOAD_ASYNC_TO_LDS for K/V tiles)
// -> out-proj GEMM (WMMA).
// All tensors (~25 MB) are L2-resident (192 MB); the kernel is matrix-pipe /
// latency bound, so the win is keeping WMMA fed: async-copy K/V one key-block
// ahead into LDS (ASYNCcnt-gated, wave-private), bias MLP on VALU co-executing
// with WMMA.
// ---------------------------------------------------------------------------

typedef __attribute__((ext_vector_type(16))) __bf16 v16bf;
typedef __attribute__((ext_vector_type(8)))  float  v8f;

#define BB   4
#define NNQ  1024
#define DD   512
#define HH   8
#define DH   64
#define HID  32
#define N3D  1536

static __device__ __forceinline__ unsigned short f32_to_bf16(float f) {
  unsigned int u = __float_as_uint(f);
  unsigned int r = u + 0x7FFFu + ((u >> 16) & 1u);   // round-to-nearest-even
  return (unsigned short)(r >> 16);
}

// A-operand (16-bit, 16x32): lane half=l/16 holds K in {8h..8h+7, 16+8h..23+8h}
// => two 16B chunks at byte offsets {16*half, 32+16*half} of the 64B row.
static __device__ __forceinline__ v16bf ld_a(const unsigned short* row, int half) {
  union { uint4 u[2]; v16bf v; } c;
  const uint4* q = (const uint4*)row;
  c.u[0] = q[half];
  c.u[1] = q[2 + half];
  return c.v;
}
// B-operand (32x16): lane = column l%16; half holds K in {8h..8h+7, 16+8h..23+8h}
// => two 16B chunks at byte offsets {16*half, 32+16*half} of the K-major row.
static __device__ __forceinline__ v16bf ld_b(const unsigned short* row, int half) {
  union { uint4 u[2]; v16bf v; } c;
  const uint4* q = (const uint4*)row;
  c.u[0] = q[half];
  c.u[1] = q[2 + half];
  return c.v;
}

static __device__ __forceinline__ v8f wmma_bf16(v16bf a, v16bf b, v8f c) {
  return __builtin_amdgcn_wmma_f32_16x16x32_bf16(false, a, false, b, (short)0, c,
                                                 false, false);
}

// Generic shared pointer -> LDS byte offset (low 32 bits of the generic addr).
static __device__ __forceinline__ unsigned lds_off(const void* p) {
  return (unsigned)(unsigned long long)p;
}

// GLOBAL_LOAD_ASYNC_TO_LDS_B128: per-lane 16B global->LDS, tracked by ASYNCcnt.
static __device__ __forceinline__ void async_load_b128(unsigned lds_addr,
                                                       const void* gaddr) {
  asm volatile("global_load_async_to_lds_b128 %0, %1, off"
               :: "v"(lds_addr), "v"(gaddr) : "memory");
}
static __device__ __forceinline__ void wait_async_le16() {
  asm volatile("s_wait_asynccnt 0x10" ::: "memory");
}
static __device__ __forceinline__ void wait_async_0() {
  asm volatile("s_wait_asynccnt 0x0" ::: "memory");
}

// --------------------------- conversion kernels ----------------------------

__global__ void cvt_cast_kernel(const float* __restrict__ src,
                                unsigned short* __restrict__ dst, int n) {
  int i = blockIdx.x * blockDim.x + threadIdx.x;
  if (i < n) dst[i] = f32_to_bf16(src[i]);
}

// src[K][Nc] (f32) -> dst[Nc][K] (bf16): B-operand wants K contiguous per col.
__global__ void cvt_transpose_kernel(const float* __restrict__ src,
                                     unsigned short* __restrict__ dst,
                                     int K, int Nc) {
  int i = blockIdx.x * blockDim.x + threadIdx.x;
  if (i < K * Nc) {
    int n = i / K, k = i - n * K;
    dst[i] = f32_to_bf16(src[k * Nc + n]);
  }
}

// ------------------------------ QKV GEMM -----------------------------------
// [4096,512]bf16 @ [512,1536] (as wT[1536,512]) -> q/k [B,H,N,dh], vT [B,H,dh,N]
__global__ __launch_bounds__(256) void qkv_gemm_kernel(
    const unsigned short* __restrict__ xb,    // [4096,512]
    const unsigned short* __restrict__ wT,    // [1536,512]
    unsigned short* __restrict__ qb,
    unsigned short* __restrict__ kb,
    unsigned short* __restrict__ vTb) {
  const int lane = threadIdx.x & 31, wave = threadIdx.x >> 5;
  const int half = lane >> 4, lanelow = lane & 15;
  const int mt = blockIdx.x & 255;          // 256 row tiles of 16
  const int sec = blockIdx.x >> 8;          // 0=Q 1=K 2=V
  const int m0 = mt * 16;
  const int n0 = sec * 512 + wave * 64;     // wave covers 64 output columns

  const unsigned short* arow = xb + (size_t)(m0 + lanelow) * DD;
  v8f acc[4] = {};
  for (int kk = 0; kk < DD; kk += 32) {
    v16bf a = ld_a(arow + kk, half);
#pragma unroll
    for (int w = 0; w < 4; ++w) {
      v16bf bm = ld_b(wT + (size_t)(n0 + w * 16 + lanelow) * DD + kk, half);
      acc[w] = wmma_bf16(a, bm, acc[w]);
    }
  }
#pragma unroll
  for (int w = 0; w < 4; ++w) {
#pragma unroll
    for (int r = 0; r < 8; ++r) {
      int m = m0 + r + 8 * half;            // global row
      int n = n0 + w * 16 + lanelow;        // global col in [0,1536)
      int b = m >> 10, i = m & 1023;
      int cc = n & 511, h = cc >> 6, d = cc & 63;
      unsigned short bv = f32_to_bf16(acc[w][r]);
      if (sec == 0)      qb[(((size_t)(b * HH + h)) * NNQ + i) * DH + d] = bv;
      else if (sec == 1) kb[(((size_t)(b * HH + h)) * NNQ + i) * DH + d] = bv;
      else               vTb[(((size_t)(b * HH + h)) * DH + d) * NNQ + i] = bv;
    }
  }
}

// ------------------------ fused bias-MLP + attention -----------------------

static __device__ __forceinline__ void bias_pair(
    float d0, float d1, float d2, bool msk,
    const float* sW1, const float* sB1, const float* sW2, const float* sB2,
    float* dst) {
  float acc[HH];
#pragma unroll
  for (int t = 0; t < HH; ++t) acc[t] = sB2[t];
  if (msk) {
#pragma unroll
    for (int t = 0; t < HH; ++t) acc[t] = -1e30f;
  } else {
#pragma unroll 4
    for (int u = 0; u < HID; ++u) {
      float hs = fmaf(d0, sW1[u],
                 fmaf(d1, sW1[HID + u],
                 fmaf(d2, sW1[2 * HID + u], sB1[u])));
      float g = 0.5f * hs * (1.0f + erff(hs * 0.70710678118654752f)); // exact GELU
#pragma unroll
      for (int t = 0; t < HH; ++t) acc[t] = fmaf(g, sW2[u * HH + t], acc[t]);
    }
  }
#pragma unroll
  for (int t = 0; t < HH; ++t) dst[t] = acc[t];
}

__global__ __launch_bounds__(256) void attn_kernel(
    const unsigned short* __restrict__ qb,   // [B,H,N,DH] bf16
    const unsigned short* __restrict__ kb,   // [B,H,N,DH] bf16
    const unsigned short* __restrict__ vTb,  // [B,H,DH,N] bf16
    const float* __restrict__ coords,        // [B,N,3]
    const unsigned char* __restrict__ mask,  // [B,N]
    const float* __restrict__ w1, const float* __restrict__ b1,
    const float* __restrict__ w2, const float* __restrict__ b2,
    unsigned short* __restrict__ attnb) {    // [B,N,D] bf16
  __shared__ float sW1[3 * HID];
  __shared__ float sB1[HID];
  __shared__ float sW2[HID * HH];
  __shared__ float sB2[HH];
  __shared__ __align__(16) float sBias[16 * 32 * HH];             // 16 KB
  __shared__ __align__(16) unsigned short sP[HH][16 * 32];        // 8 KB
  // Double-buffered K/V tiles per head (wave-private, ASYNCcnt-gated):
  __shared__ __align__(16) unsigned short sK[HH][2][32 * DH];     // 64 KB
  __shared__ __align__(16) unsigned short sV[HH][2][DH * 32];     // 64 KB

  const int tid = threadIdx.x;
  const int lane = tid & 31, h = tid >> 5;     // wave == head
  const int half = lane >> 4, lanelow = lane & 15;
  const int b = blockIdx.x >> 6;
  const int itile = (blockIdx.x & 63) * 16;

  if (tid < 3 * HID) sW1[tid] = w1[tid];
  if (tid < HID)     sB1[tid] = b1[tid];
  if (tid < HID * HH) sW2[tid] = w2[tid];      // 256 entries == blockDim
  if (tid < HH)      sB2[tid] = b2[tid];
  __syncthreads();

  const unsigned short* kbase = kb + ((size_t)(b * HH + h)) * NNQ * DH;
  const unsigned short* vbase = vTb + ((size_t)(b * HH + h)) * DH * NNQ;

  // ---- async staging of one 32-key block (16 x b128-per-lane issues) ----
  auto issue_kv = [&](int jnext, int buf) {
    // K tile: rows jnext..jnext+31 of [N,DH] are one contiguous 4 KB blob.
    const char* kg = (const char*)(kbase + (size_t)jnext * DH);
    unsigned skb = lds_off(&sK[h][buf][0]);
#pragma unroll
    for (int t = 0; t < 8; ++t) {
      int c = t * 32 + lane;                       // 16B chunk 0..255
      async_load_b128(skb + c * 16, kg + c * 16);
    }
    // V tile: 64 rows (dh) x 64 B, row stride NNQ*2 bytes in vT.
    const char* vg = (const char*)(vbase + jnext);
    unsigned svb = lds_off(&sV[h][buf][0]);
#pragma unroll
    for (int t = 0; t < 8; ++t) {
      int c = t * 32 + lane;                       // 0..255
      int row = c >> 2, ch = c & 3;                // 4 chunks per 64B row
      async_load_b128(svb + row * 64 + ch * 16,
                      vg + (size_t)row * (NNQ * 2) + ch * 16);
    }
  };
  issue_kv(0, 0);                                  // prologue prefetch

  // Preload Q (16 rows x 64 dh) into two A-operands.
  const unsigned short* qrow =
      qb + (((size_t)(b * HH + h)) * NNQ + itile + lanelow) * DH;
  v16bf qa0 = ld_a(qrow, half);
  v16bf qa1 = ld_a(qrow + 32, half);

  v8f o0 = {}, o1 = {}, o2 = {}, o3 = {};
  float mrun[8], lrun[8];
#pragma unroll
  for (int r = 0; r < 8; ++r) { mrun[r] = -INFINITY; lrun[r] = 0.f; }

  // Bias-MLP pair assignment: thread -> (iA=tid/32, j=tid%32) and (iA+8, j).
  const int ia = tid >> 5, bj = tid & 31;
  const float* cA = coords + (size_t)(b * NNQ + itile + ia) * 3;
  const float* cBp = coords + (size_t)(b * NNQ + itile + 8 + ia) * 3;
  const float ax = cA[0], ay = cA[1], az = cA[2];
  const float bx = cBp[0], by = cBp[1], bz = cBp[2];

  const float scale = 0.125f;                      // dh^-0.5 = 64^-0.5

  for (int jtile = 0; jtile < NNQ; jtile += 32) {
    const int buf = (jtile >> 5) & 1;

    // ---- prefetch next block, then retire this block's async copies ----
    if (jtile + 32 < NNQ) {
      issue_kv(jtile + 32, buf ^ 1);
      wait_async_le16();        // 16 newest outstanding -> current tile done
    } else {
      wait_async_0();
    }

    // ---- bias tile: 256 threads x 2 pairs -> sBias[16][32][8] ----
    {
      const float* cj = coords + (size_t)(b * NNQ + jtile + bj) * 3;
      float jx = cj[0], jy = cj[1], jz = cj[2];
      bool msk = mask[b * NNQ + jtile + bj] != 0;
      bias_pair(ax - jx, ay - jy, az - jz, msk, sW1, sB1, sW2, sB2,
                &sBias[(ia * 32 + bj) * HH]);
      bias_pair(bx - jx, by - jy, bz - jz, msk, sW1, sB1, sW2, sB2,
                &sBias[((ia + 8) * 32 + bj) * HH]);
    }

    // ---- S = Q K^T from LDS-staged K (4 WMMA) ----
    const unsigned short* kr0 = &sK[h][buf][(size_t)lanelow * DH];
    const unsigned short* kr1 = &sK[h][buf][(size_t)(16 + lanelow) * DH];
    v8f s0 = {}, s1 = {};
    s0 = wmma_bf16(qa0, ld_b(kr0, half), s0);
    s0 = wmma_bf16(qa1, ld_b(kr0 + 32, half), s0);
    s1 = wmma_bf16(qa0, ld_b(kr1, half), s1);
    s1 = wmma_bf16(qa1, ld_b(kr1 + 32, half), s1);

    __syncthreads();   // bias tile visible to all waves

    // ---- online softmax (C-layout: r -> row r+8*half, col lanelow) ----
#pragma unroll
    for (int r = 0; r < 8; ++r) {
      int il = r + 8 * half;
      float v0 = s0[r] * scale + sBias[(il * 32 + lanelow) * HH + h];
      float v1 = s1[r] * scale + sBias[(il * 32 + 16 + lanelow) * HH + h];
      float mx = fmaxf(v0, v1);
#pragma unroll
      for (int mk = 1; mk < 16; mk <<= 1) mx = fmaxf(mx, __shfl_xor(mx, mk, 32));
      float mn = fmaxf(mrun[r], mx);
      float alpha = __expf(mrun[r] - mn);
      float p0 = __expf(v0 - mn);
      float p1 = __expf(v1 - mn);
      float ps = p0 + p1;
#pragma unroll
      for (int mk = 1; mk < 16; mk <<= 1) ps += __shfl_xor(ps, mk, 32);
      lrun[r] = lrun[r] * alpha + ps;
      mrun[r] = mn;
      o0[r] *= alpha; o1[r] *= alpha; o2[r] *= alpha; o3[r] *= alpha;
      sP[h][il * 32 + lanelow] = f32_to_bf16(p0);        // wave-private tile
      sP[h][il * 32 + 16 + lanelow] = f32_to_bf16(p1);
    }
    __syncthreads();   // bias reads done (region reused next iter) + P in LDS

    // ---- O += P V from LDS-staged V (4 WMMA over dh) ----
    v16bf pa = ld_a(&sP[h][lanelow * 32], half);
    o0 = wmma_bf16(pa, ld_b(&sV[h][buf][(0 * 16 + lanelow) * 32], half), o0);
    o1 = wmma_bf16(pa, ld_b(&sV[h][buf][(1 * 16 + lanelow) * 32], half), o1);
    o2 = wmma_bf16(pa, ld_b(&sV[h][buf][(2 * 16 + lanelow) * 32], half), o2);
    o3 = wmma_bf16(pa, ld_b(&sV[h][buf][(3 * 16 + lanelow) * 32], half), o3);
  }

  // ---- normalize, store as bf16 into [B,N,D] (head-major columns) ----
#pragma unroll
  for (int r = 0; r < 8; ++r) {
    int i = itile + r + 8 * half;
    float inv = (lrun[r] > 0.f) ? 1.f / lrun[r] : 0.f;
    unsigned short* orow = attnb + ((size_t)(b * NNQ + i)) * DD + h * DH;
    orow[0 + lanelow]  = f32_to_bf16(o0[r] * inv);
    orow[16 + lanelow] = f32_to_bf16(o1[r] * inv);
    orow[32 + lanelow] = f32_to_bf16(o2[r] * inv);
    orow[48 + lanelow] = f32_to_bf16(o3[r] * inv);
  }
}

// ------------------------------ out-proj GEMM ------------------------------
__global__ __launch_bounds__(256) void outproj_kernel(
    const unsigned short* __restrict__ attnb,  // [4096,512]
    const unsigned short* __restrict__ woT,    // [512,512] transposed
    const float* __restrict__ outb,            // [512]
    float* __restrict__ out) {                 // [4096,512]
  const int lane = threadIdx.x & 31, wave = threadIdx.x >> 5;
  const int half = lane >> 4, lanelow = lane & 15;
  const int m0 = blockIdx.x * 16;
  const int n0 = wave * 64;

  const unsigned short* arow = attnb + (size_t)(m0 + lanelow) * DD;
  v8f acc[4] = {};
  for (int kk = 0; kk < DD; kk += 32) {
    v16bf a = ld_a(arow + kk, half);
#pragma unroll
    for (int w = 0; w < 4; ++w) {
      v16bf bm = ld_b(woT + (size_t)(n0 + w * 16 + lanelow) * DD + kk, half);
      acc[w] = wmma_bf16(a, bm, acc[w]);
    }
  }
#pragma unroll
  for (int w = 0; w < 4; ++w) {
#pragma unroll
    for (int r = 0; r < 8; ++r) {
      int m = m0 + r + 8 * half;
      int n = n0 + w * 16 + lanelow;
      out[(size_t)m * DD + n] = acc[w][r] + outb[n];
    }
  }
}

// ------------------------------- launcher ----------------------------------
extern "C" void kernel_launch(void* const* d_in, const int* in_sizes, int n_in,
                              void* d_out, int out_size, void* d_ws, size_t ws_size,
                              hipStream_t stream) {
  (void)in_sizes; (void)n_in; (void)out_size; (void)ws_size;
  const float* x      = (const float*)d_in[0];
  const float* coords = (const float*)d_in[1];
  const unsigned char* mask = (const unsigned char*)d_in[2];
  const float* qkv_w  = (const float*)d_in[3];
  const float* out_w  = (const float*)d_in[4];
  const float* out_b  = (const float*)d_in[5];
  const float* w1     = (const float*)d_in[6];
  const float* b1     = (const float*)d_in[7];
  const float* w2     = (const float*)d_in[8];
  const float* b2     = (const float*)d_in[9];
  float* out = (float*)d_out;

  // Workspace layout (bytes); total 22.0 MB.
  char* ws = (char*)d_ws;
  unsigned short* xb    = (unsigned short*)(ws + 0);          // 4 MB  x bf16
  unsigned short* wqkvT = (unsigned short*)(ws + 4194304);    // 1.5MB qkv_w^T
  unsigned short* woT   = (unsigned short*)(ws + 5767168);    // 0.5MB out_w^T
  unsigned short* qb    = (unsigned short*)(ws + 6291456);    // 4 MB  q
  unsigned short* kbuf  = (unsigned short*)(ws + 10485760);   // 4 MB  k
  unsigned short* vTb   = (unsigned short*)(ws + 14680064);   // 4 MB  v^T
  unsigned short* attnb = (unsigned short*)(ws + 18874368);   // 4 MB  attn out

  cvt_cast_kernel<<<(BB * NNQ * DD + 255) / 256, 256, 0, stream>>>(x, xb, BB * NNQ * DD);
  cvt_transpose_kernel<<<(DD * N3D + 255) / 256, 256, 0, stream>>>(qkv_w, wqkvT, DD, N3D);
  cvt_transpose_kernel<<<(DD * DD + 255) / 256, 256, 0, stream>>>(out_w, woT, DD, DD);
  qkv_gemm_kernel<<<256 * 3, 256, 0, stream>>>(xb, wqkvT, qb, kbuf, vTb);
  attn_kernel<<<BB * (NNQ / 16), 256, 0, stream>>>(qb, kbuf, vTb, coords, mask,
                                                   w1, b1, w2, b2, attnb);
  outproj_kernel<<<BB * NNQ / 16, 256, 0, stream>>>(attnb, woT, out_b, out);
}
